// NodeEmbedder_88785563943710
// MI455X (gfx1250) — matrix operand order repaired and checked by reading, not compiled, not run on hardware
//
#include <hip/hip_runtime.h>
#include <hip/hip_bf16.h>

typedef __bf16 bf16_t;
typedef __attribute__((ext_vector_type(16))) __bf16 v16bf;
typedef __attribute__((ext_vector_type(8)))  __bf16 v8bf;
typedef __attribute__((ext_vector_type(8)))  float  v8f;
typedef __attribute__((ext_vector_type(4)))  int    v4i;

#define F 128
#define LDSF_STRIDE 132   // 128 + 4 pad floats (528B rows: multiple of 16B)

// -------- async global->LDS path (guarded; falls back to load+store staging) --
#if defined(__has_builtin)
#  if __has_builtin(__builtin_amdgcn_global_load_async_to_lds_b128)
#    define HAVE_ASYNC_LDS 1
#  endif
#  if __has_builtin(__builtin_amdgcn_s_wait_asynccnt)
#    define HAVE_WAIT_ASYNC_BUILTIN 1
#  endif
#endif

// Builtin signature (from hipcc diagnostic): param0 = v4i* in AS1 (global),
// param1 = v4i* in AS3 (LDS), then two immediate ints (offset, cpol).
typedef __attribute__((address_space(1))) v4i* gv4i_p;   // pointer to global v4i
typedef __attribute__((address_space(3))) v4i* lv4i_p;   // pointer to LDS v4i

__device__ __forceinline__ void wait_async0() {
#if defined(HAVE_WAIT_ASYNC_BUILTIN)
    __builtin_amdgcn_s_wait_asynccnt(0);
#else
    asm volatile("s_wait_asynccnt 0" ::: "memory");
#endif
}

// Stage one 128x128 f32 tile of A into LDS (rows >= M zero-filled).
__device__ __forceinline__ void stage_A_tile(
    const float* __restrict__ A, float (*Asf)[LDSF_STRIDE],
    int rowbase, int M, int tid)
{
#if defined(HAVE_ASYNC_LDS)
    for (int i = tid; i < 128 * 32; i += 256) {
        int r  = i >> 5;
        int c4 = (i & 31) << 2;
        int gr = rowbase + r;
        if (gr < M) {
            float* gp = const_cast<float*>(A) + (size_t)gr * F + c4;
            __builtin_amdgcn_global_load_async_to_lds_b128(
                (gv4i_p)gp, (lv4i_p)(&Asf[r][c4]), 0, 0);
        } else {
            *reinterpret_cast<float4*>(&Asf[r][c4]) = make_float4(0.f, 0.f, 0.f, 0.f);
        }
    }
    wait_async0();
#else
    for (int i = tid; i < 128 * 32; i += 256) {
        int r  = i >> 5;
        int c4 = (i & 31) << 2;
        int gr = rowbase + r;
        float4 v = make_float4(0.f, 0.f, 0.f, 0.f);
        if (gr < M) v = *reinterpret_cast<const float4*>(A + (size_t)gr * F + c4);
        *reinterpret_cast<float4*>(&Asf[r][c4]) = v;
    }
#endif
    __syncthreads();
}

// Build one v16bf A fragment (16-bit A-matrix ISA layout) from the f32 LDS tile.
//   K = kk*32 + 8*(lane>>4) + (e < 8 ? e : e + 8)
__device__ __forceinline__ v16bf make_afrag(
    const float (*Asf)[LDSF_STRIDE], int w, int lr, int lhi, int kk)
{
    const float* ap = &Asf[w * 16 + lr][kk * 32 + lhi * 8];
    float4 a0 = *reinterpret_cast<const float4*>(ap);
    float4 a1 = *reinterpret_cast<const float4*>(ap + 4);
    float4 a2 = *reinterpret_cast<const float4*>(ap + 16);
    float4 a3 = *reinterpret_cast<const float4*>(ap + 20);
    v16bf af;
    af[0]  = (bf16_t)a0.x; af[1]  = (bf16_t)a0.y; af[2]  = (bf16_t)a0.z; af[3]  = (bf16_t)a0.w;
    af[4]  = (bf16_t)a1.x; af[5]  = (bf16_t)a1.y; af[6]  = (bf16_t)a1.z; af[7]  = (bf16_t)a1.w;
    af[8]  = (bf16_t)a2.x; af[9]  = (bf16_t)a2.y; af[10] = (bf16_t)a2.z; af[11] = (bf16_t)a2.w;
    af[12] = (bf16_t)a3.x; af[13] = (bf16_t)a3.y; af[14] = (bf16_t)a3.z; af[15] = (bf16_t)a3.w;
    return af;
}

__device__ __forceinline__ v16bf load_bfrag(const bf16_t* __restrict__ bp)
{
    v8bf blo = *reinterpret_cast<const v8bf*>(bp);
    v8bf bhi = *reinterpret_cast<const v8bf*>(bp + 8);
    v16bf bfr;
    #pragma unroll
    for (int e = 0; e < 8; ++e) { bfr[e] = blo[e]; bfr[e + 8] = bhi[e]; }
    return bfr;
}

// ---------------------------------------------------------------------------
// Weight swizzle: f32 [K=128][N=128] row-major -> bf16 fragment-ready layout
//   out[((kk*8 + nt)*32 + lane)*16 + e] = W[kk*32 + (lane>>4)*8 + (e < 8 ? e : e + 8)][nt*16 + (lane&15)]
// 14 blocks: 0=preW0, 1=preW1, 2..7=ginW, 8..12=postW0 (5 K-blocks), 13=postW1
// ---------------------------------------------------------------------------
__global__ __launch_bounds__(256) void k_swizzle_w(
    const float* __restrict__ preW0, const float* __restrict__ preW1,
    const float* __restrict__ ginW,  const float* __restrict__ postW0,
    const float* __restrict__ postW1, bf16_t* __restrict__ out)
{
    const int b = blockIdx.x;
    const float* W;
    if      (b == 0) W = preW0;
    else if (b == 1) W = preW1;
    else if (b < 8)  W = ginW  + (b - 2) * (F * F);
    else if (b < 13) W = postW0 + (b - 8) * (F * F);
    else             W = postW1;
    bf16_t* o = out + (size_t)b * (F * F);
    for (int i = threadIdx.x; i < F * F; i += 256) {
        int e  = i & 15;
        int l  = (i >> 4) & 31;
        int t  = i >> 9;            // kk*8 + nt
        int kk = t >> 3, nt = t & 7;
        int K  = kk * 32 + ((l >> 4) << 3) + (e < 8 ? e : e + 8);
        int n  = nt * 16 + (l & 15);
        o[i] = (bf16_t)W[K * F + n];
    }
}

// ---------------------------------------------------------------------------
// GEMM C[M,128] = epi(A[M,128] @ W[128,128] + bias)
// epi: 0 = bias, 1 = bias+relu, 2 = bias+FiLM(gamma/beta gathered via batch_vec)
// ---------------------------------------------------------------------------
__global__ __launch_bounds__(256, 1) void k_gemm128(
    const float* __restrict__ A, const bf16_t* __restrict__ Wsw,
    const float* __restrict__ bias, float* __restrict__ C, int M, int epi,
    const float* __restrict__ gam, const float* __restrict__ bet,
    const int* __restrict__ batch, int conv)
{
    __shared__ __align__(16) float Asf[F][LDSF_STRIDE];
    const int tid = threadIdx.x;
    const int rowbase = blockIdx.x * 128;

    stage_A_tile(A, Asf, rowbase, M, tid);

    const int w    = tid >> 5;
    const int lane = tid & 31;
    const int lhi  = lane >> 4;
    const int lr   = lane & 15;

    v8f acc[8] = {};

    #pragma unroll
    for (int kk = 0; kk < 4; ++kk) {
        v16bf af = make_afrag(Asf, w, lr, lhi, kk);
        #pragma unroll
        for (int nt = 0; nt < 8; ++nt) {
            const bf16_t* bp = Wsw + (((kk << 3) + nt) * 32 + lane) * 16;
            v16bf bfr = load_bfrag(bp);
            acc[nt] = __builtin_amdgcn_wmma_f32_16x16x32_bf16(
                false, af, false, bfr, (short)0, acc[nt], false, false);
        }
    }

    // C/D layout: lane -> N = lane&15, VGPR r -> M = r + 8*(lane>>4)
    #pragma unroll
    for (int nt = 0; nt < 8; ++nt) {
        const int nc = nt * 16 + lr;
        const float bv = bias[nc];
        #pragma unroll
        for (int r = 0; r < 8; ++r) {
            const int row = rowbase + w * 16 + lhi * 8 + r;
            if (row < M) {
                float v = acc[nt][r] + bv;
                if (epi == 1) {
                    v = fmaxf(v, 0.f);
                } else if (epi == 2) {
                    const int g = batch[row];
                    v = v * gam[g * 3 + conv] + bet[g * 3 + conv];
                }
                C[(size_t)row * F + nc] = v;
            }
        }
    }
}

// ---------------------------------------------------------------------------
// GEMM over concatenated K=640: C = sum_b A_b @ W0_b + bias  (jumping knowledge)
// ---------------------------------------------------------------------------
__global__ __launch_bounds__(256, 1) void k_gemm640(
    const float* __restrict__ A0, const float* __restrict__ A1,
    const float* __restrict__ A2, const float* __restrict__ A3,
    const float* __restrict__ A4, const bf16_t* __restrict__ Wsw,
    const float* __restrict__ bias, float* __restrict__ C, int M)
{
    __shared__ __align__(16) float Asf[F][LDSF_STRIDE];
    const int tid = threadIdx.x;
    const int rowbase = blockIdx.x * 128;
    const int w    = tid >> 5;
    const int lane = tid & 31;
    const int lhi  = lane >> 4;
    const int lr   = lane & 15;

    v8f acc[8] = {};
    const float* Aptr[5] = {A0, A1, A2, A3, A4};

    #pragma unroll
    for (int ab = 0; ab < 5; ++ab) {
        const float* A = Aptr[ab];
        __syncthreads();   // previous iteration's LDS reads complete
        stage_A_tile(A, Asf, rowbase, M, tid);

        #pragma unroll
        for (int kk = 0; kk < 4; ++kk) {
            v16bf af = make_afrag(Asf, w, lr, lhi, kk);
            #pragma unroll
            for (int nt = 0; nt < 8; ++nt) {
                const bf16_t* bp = Wsw + (size_t)((ab * 4 + kk) * 8 + nt) * 512 + lane * 16;
                v16bf bfr = load_bfrag(bp);
                acc[nt] = __builtin_amdgcn_wmma_f32_16x16x32_bf16(
                    false, af, false, bfr, (short)0, acc[nt], false, false);
            }
        }
    }

    #pragma unroll
    for (int nt = 0; nt < 8; ++nt) {
        const int nc = nt * 16 + lr;
        const float bv = bias[nc];
        #pragma unroll
        for (int r = 0; r < 8; ++r) {
            const int row = rowbase + w * 16 + lhi * 8 + r;
            if (row < M) C[(size_t)row * F + nc] = acc[nt][r] + bv;
        }
    }
}

// ---------------------------------------------------------------------------
// agg init: agg = h (GIN eps=0 self term), float4 copy
// ---------------------------------------------------------------------------
__global__ __launch_bounds__(256) void k_copy4(
    const float* __restrict__ src, float* __restrict__ dst, int n4)
{
    int i = blockIdx.x * 256 + threadIdx.x;
    if (i < n4)
        reinterpret_cast<float4*>(dst)[i] = reinterpret_cast<const float4*>(src)[i];
}

// ---------------------------------------------------------------------------
// Edge scatter-add: agg[dst] += h[src]. One wave per edge, float4 per lane.
// h is L2-resident (51 MB < 192 MB), atomics land in L2.
// ---------------------------------------------------------------------------
__global__ __launch_bounds__(256) void k_scatter(
    const int* __restrict__ srcI, const int* __restrict__ dstI,
    const float* __restrict__ h, float* __restrict__ agg, int E)
{
    const int e    = (blockIdx.x * 256 + threadIdx.x) >> 5;
    const int lane = threadIdx.x & 31;
    if (e >= E) return;
    const int s = srcI[e];
    const int d = dstI[e];
    float4 v = reinterpret_cast<const float4*>(h + (size_t)s * F)[lane];
    float* p = agg + (size_t)d * F + lane * 4;
    atomicAdd(p + 0, v.x);
    atomicAdd(p + 1, v.y);
    atomicAdd(p + 2, v.z);
    atomicAdd(p + 3, v.w);
}

// ---------------------------------------------------------------------------
// out = relu(layernorm(z) * g + b); one wave per 128-elem row
// ---------------------------------------------------------------------------
__global__ __launch_bounds__(256) void k_lnrelu(
    const float* __restrict__ z, float* __restrict__ out,
    const float* __restrict__ g, const float* __restrict__ b, int M)
{
    const int row  = (blockIdx.x * 256 + threadIdx.x) >> 5;
    const int lane = threadIdx.x & 31;
    if (row >= M) return;
    float4 v = reinterpret_cast<const float4*>(z + (size_t)row * F)[lane];
    float s  = v.x + v.y + v.z + v.w;
    float s2 = v.x * v.x + v.y * v.y + v.z * v.z + v.w * v.w;
    #pragma unroll
    for (int off = 16; off > 0; off >>= 1) {
        s  += __shfl_xor(s,  off, 32);
        s2 += __shfl_xor(s2, off, 32);
    }
    const float mu  = s * (1.f / 128.f);
    const float var = s2 * (1.f / 128.f) - mu * mu;
    const float inv = rsqrtf(var + 1e-5f);
    float4 gg = reinterpret_cast<const float4*>(g)[lane];
    float4 bb = reinterpret_cast<const float4*>(b)[lane];
    float4 o;
    o.x = fmaxf(0.f, (v.x - mu) * inv * gg.x + bb.x);
    o.y = fmaxf(0.f, (v.y - mu) * inv * gg.y + bb.y);
    o.z = fmaxf(0.f, (v.z - mu) * inv * gg.z + bb.z);
    o.w = fmaxf(0.f, (v.w - mu) * inv * gg.w + bb.w);
    reinterpret_cast<float4*>(out + (size_t)row * F)[lane] = o;
}

// ---------------------------------------------------------------------------
extern "C" void kernel_launch(void* const* d_in, const int* in_sizes, int n_in,
                              void* d_out, int out_size, void* d_ws, size_t ws_size,
                              hipStream_t stream)
{
    const float* x      = (const float*)d_in[0];
    const int*   edge   = (const int*)  d_in[1];
    const int*   batch  = (const int*)  d_in[2];
    const float* gammas = (const float*)d_in[3];
    const float* betas  = (const float*)d_in[4];
    const float* preW0  = (const float*)d_in[5];
    const float* preb0  = (const float*)d_in[6];
    const float* preW1  = (const float*)d_in[7];
    const float* preb1  = (const float*)d_in[8];
    const float* ginW   = (const float*)d_in[9];
    const float* ginb   = (const float*)d_in[10];
    const float* postW0 = (const float*)d_in[11];
    const float* postb0 = (const float*)d_in[12];
    const float* lng    = (const float*)d_in[13];
    const float* lnb    = (const float*)d_in[14];
    const float* postW1 = (const float*)d_in[15];
    const float* postb1 = (const float*)d_in[16];

    const int N = in_sizes[0] / F;        // 100000
    const int E = in_sizes[1] / 2;        // 1600000
    float* out = (float*)d_out;

    // Workspace: 4 jump buffers + 1 agg buffer (f32 [N,128]) + swizzled weights
    const size_t bufElems = (size_t)N * F;
    float* J1  = (float*)d_ws;
    float* J2  = J1 + bufElems;
    float* J3  = J2 + bufElems;
    float* J4  = J3 + bufElems;
    float* agg = J4 + bufElems;
    bf16_t* sw = (bf16_t*)(agg + bufElems);
    float* hid = out;                     // d_out doubles as MLP-hidden scratch

    const int MB  = (N + 127) / 128;                   // 782 GEMM row-tiles
    const int CB  = ((int)(bufElems / 4) + 255) / 256; // float4 copy blocks
    const int SB  = ((E + 7) / 8);                     // 8 edges per 256-thr block
    const int LB  = (N + 7) / 8;                       // 8 rows per LN block
    const size_t WMAT = (size_t)F * F;                 // 16384 elems per matrix

    k_swizzle_w<<<14, 256, 0, stream>>>(preW0, preW1, ginW, postW0, postW1, sw);

    // preprocess MLP
    k_gemm128<<<MB, 256, 0, stream>>>(x,   sw,         preb0, hid, N, 1, nullptr, nullptr, nullptr, 0);
    k_gemm128<<<MB, 256, 0, stream>>>(hid, sw + WMAT,  preb1, J1,  N, 0, nullptr, nullptr, nullptr, 0);

    // 3 GIN convs with FiLM
    float* Jarr[4] = {J1, J2, J3, J4};
    for (int i = 0; i < 3; ++i) {
        const float* hin  = Jarr[i];
        float*       hout = Jarr[i + 1];
        k_copy4  <<<CB, 256, 0, stream>>>(hin, agg, (int)(bufElems / 4));
        k_scatter<<<SB, 256, 0, stream>>>(edge, edge + E, hin, agg, E);
        k_gemm128<<<MB, 256, 0, stream>>>(agg, sw + (size_t)(2 + 2 * i) * WMAT,
                                          ginb + (2 * i) * F, hid, N, 1,
                                          nullptr, nullptr, nullptr, 0);
        k_gemm128<<<MB, 256, 0, stream>>>(hid, sw + (size_t)(3 + 2 * i) * WMAT,
                                          ginb + (2 * i + 1) * F, hout, N, 2,
                                          gammas, betas, batch, i);
    }

    // postprocess: jumping-knowledge GEMM (K=640), LN+ReLU, final GEMM
    k_gemm640<<<MB, 256, 0, stream>>>(x, J1, J2, J3, J4, sw + 8 * WMAT, postb0, agg, N);
    k_lnrelu <<<LB, 256, 0, stream>>>(agg, hid, lng, lnb, N);
    k_gemm128<<<MB, 256, 0, stream>>>(hid, sw + 13 * WMAT, postb1, out, N, 0,
                                      nullptr, nullptr, nullptr, 0);
}
